// GCN_996432413323
// MI455X (gfx1250) — matrix operand order repaired and checked by reading, not compiled, not run on hardware
//
#include <hip/hip_runtime.h>
#include <stdint.h>

#define NROWS 16384
#define FEAT  512
#define HID   512
#define NOUTC 128

typedef __attribute__((ext_vector_type(16))) __bf16 v16bf;
typedef __attribute__((ext_vector_type(8)))  __bf16 bf16x8;
typedef __attribute__((ext_vector_type(8)))  float  v8f;

__device__ __forceinline__ unsigned short f2bf(float x) {
  union { float f; unsigned u; } v; v.f = x;
  unsigned r = v.u + 0x7FFFu + ((v.u >> 16) & 1u); // round-to-nearest-even
  return (unsigned short)(r >> 16);
}

union FragAB { v16bf v; bf16x8 h[2]; };
union Pack8  { unsigned short s[8]; uint4 q; };

// gfx1250 async global->LDS copy, 16 bytes per lane; tracked by ASYNCcnt.
__device__ __forceinline__ void async_copy16(unsigned lds_addr, const void* base, unsigned byte_off) {
  asm volatile("global_load_async_to_lds_b128 %0, %1, %2"
               :: "v"(lds_addr), "v"(byte_off), "s"(base) : "memory");
}
__device__ __forceinline__ void wait_async0() {
  asm volatile("s_wait_asynccnt 0x0" ::: "memory");
}

// One K-step (K=32) of the 128x128 block: 2x4 WMMA tiles per wave.
__device__ __forceinline__ void compute_step(const unsigned short* sA, const unsigned short* sB,
                                             int wm, int wn, int r, int half, v8f (&acc)[2][4]) {
  constexpr int LD = 40;
  FragAB a[2], b[4];
#pragma unroll
  for (int i = 0; i < 2; ++i) {
    const int row = wm + i * 16 + r;
    // A 16x32 bf16: lane holds row; K-runs at half*8 and half*8+16
    a[i].h[0] = *(const bf16x8*)&sA[row * LD + half * 8];
    a[i].h[1] = *(const bf16x8*)&sA[row * LD + half * 8 + 16];
  }
#pragma unroll
  for (int j = 0; j < 4; ++j) {
    const int nn = wn + j * 16 + r;
    // B 32x16 bf16: lane holds column; K-runs at half*8 and half*8+16
    b[j].h[0] = *(const bf16x8*)&sB[nn * LD + half * 8];
    b[j].h[1] = *(const bf16x8*)&sB[nn * LD + half * 8 + 16];
  }
#pragma unroll
  for (int i = 0; i < 2; ++i)
#pragma unroll
    for (int j = 0; j < 4; ++j)
      acc[i][j] = __builtin_amdgcn_wmma_f32_16x16x32_bf16(
          false, a[i].v, false, b[j].v, (short)0, acc[i][j], false, false);
}

// MODE 0: out = bf16 [Nc x M] transposed, no bias/relu
// MODE 1: out = bf16 [M x Nc] row-major, + bias, relu
// MODE 2: out = f32  [M x Nc] row-major, + bias
template<int MODE>
__device__ __forceinline__ void epilogue(v8f (&acc)[2][4], const float* bias, void* outv,
                                         int M, int Nc, int m0, int n0, int wm, int wn,
                                         int r, int half) {
#pragma unroll
  for (int i = 0; i < 2; ++i) {
    const int mg = m0 + wm + i * 16 + half * 8; // C: VGPR v -> row v+8*half, col lane%16
#pragma unroll
    for (int j = 0; j < 4; ++j) {
      const int ng = n0 + wn + j * 16 + r;
      if constexpr (MODE == 0) {
        unsigned short* outT = (unsigned short*)outv;
        Pack8 p;
#pragma unroll
        for (int v = 0; v < 8; ++v) p.s[v] = f2bf(acc[i][j][v]);
        *(uint4*)&outT[(size_t)ng * M + mg] = p.q;
      } else if constexpr (MODE == 1) {
        unsigned short* out = (unsigned short*)outv;
        const float bv = bias[ng];
#pragma unroll
        for (int v = 0; v < 8; ++v) {
          float val = acc[i][j][v] + bv;
          val = val > 0.f ? val : 0.f;
          out[(size_t)(mg + v) * Nc + ng] = f2bf(val);
        }
      } else {
        float* out = (float*)outv;
        const float bv = bias[ng];
#pragma unroll
        for (int v = 0; v < 8; ++v)
          out[(size_t)(mg + v) * Nc + ng] = acc[i][j][v] + bv;
      }
    }
  }
}

// C[M x Nc] = A[M x K] (bf16) * B (bf16, given transposed Bt[Nc x K]).
// Double-buffered LDS, tiles staged with async global->LDS copies (no VGPR staging).
template<int MODE>
__global__ __launch_bounds__(256, 1)
void gemm_wmma_async(const unsigned short* __restrict__ A, const unsigned short* __restrict__ Bt,
                     const float* __restrict__ bias, void* __restrict__ outv,
                     int M, int Nc, int K)
{
  constexpr int BM = 128, BN = 128, KB = 32, LD = 40;
  __shared__ alignas(16) unsigned short smA[2][BM * LD];
  __shared__ alignas(16) unsigned short smB[2][BN * LD];

  const int tid  = threadIdx.x;
  const int m0   = blockIdx.x * BM;
  const int n0   = blockIdx.y * BN;
  const int wid  = tid >> 5, lane = tid & 31;
  const int r    = lane & 15, half = lane >> 4;
  const int wm   = (wid & 3) * 32, wn = (wid >> 2) * 64;

  // Each thread async-copies 4x16B per K-tile: rows rr and rr+64, 8 bf16 columns.
  const int rr = tid >> 2;
  const int c8 = (tid & 3) * 8;
  const unsigned aoff0 = (unsigned)(((size_t)(m0 + rr) * K + c8) * 2);
  const unsigned aoff1 = (unsigned)(((size_t)(m0 + rr + 64) * K + c8) * 2);
  const unsigned boff0 = (unsigned)(((size_t)(n0 + rr) * K + c8) * 2);
  const unsigned boff1 = (unsigned)(((size_t)(n0 + rr + 64) * K + c8) * 2);
  const unsigned ldsA0 = (unsigned)(uintptr_t)&smA[0][rr * LD + c8];
  const unsigned ldsA1 = (unsigned)(uintptr_t)&smA[0][(rr + 64) * LD + c8];
  const unsigned ldsB0 = (unsigned)(uintptr_t)&smB[0][rr * LD + c8];
  const unsigned ldsB1 = (unsigned)(uintptr_t)&smB[0][(rr + 64) * LD + c8];
  constexpr unsigned BUFSZ = (unsigned)(BM * LD * 2); // bytes per buffer

  const int nk = K / KB;

  // prefetch K-tile 0 into buffer 0
  async_copy16(ldsA0, A, aoff0);  async_copy16(ldsA1, A, aoff1);
  async_copy16(ldsB0, Bt, boff0); async_copy16(ldsB1, Bt, boff1);

  v8f acc[2][4] = {};
  for (int kt = 0; kt < nk; ++kt) {
    wait_async0();       // this wave's copies into buf kt are done
    __syncthreads();     // everyone's copies visible; everyone done reading other buf
    if (kt + 1 < nk) {   // refill the other buffer; overlaps with compute below
      const unsigned kb = (unsigned)((kt + 1) * KB * 2);
      const unsigned lb = ((kt + 1) & 1) ? BUFSZ : 0u;
      async_copy16(ldsA0 + lb, A, aoff0 + kb);  async_copy16(ldsA1 + lb, A, aoff1 + kb);
      async_copy16(ldsB0 + lb, Bt, boff0 + kb); async_copy16(ldsB1 + lb, Bt, boff1 + kb);
    }
    const int buf = kt & 1;
    compute_step(smA[buf], smB[buf], wm, wn, r, half, acc);
  }
  epilogue<MODE>(acc, bias, outv, M, Nc, m0, n0, wm, wn, r, half);
}

// Fallback when workspace can't hold bf16(adj): A is f32, converted on the fly.
// Single-buffered (two barriers per K-step) so no long live ranges -> no spills.
template<int MODE>
__global__ __launch_bounds__(256, 1)
void gemm_wmma_f32A(const float* __restrict__ A, const unsigned short* __restrict__ Bt,
                    const float* __restrict__ bias, void* __restrict__ outv,
                    int M, int Nc, int K)
{
  constexpr int BM = 128, BN = 128, KB = 32, LD = 40;
  __shared__ alignas(16) unsigned short smA[BM * LD];
  __shared__ alignas(16) unsigned short smB[BN * LD];

  const int tid  = threadIdx.x;
  const int m0   = blockIdx.x * BM;
  const int n0   = blockIdx.y * BN;
  const int wid  = tid >> 5, lane = tid & 31;
  const int r    = lane & 15, half = lane >> 4;
  const int wm   = (wid & 3) * 32, wn = (wid >> 2) * 64;

  v8f acc[2][4] = {};
  const int nk = K / KB;
  for (int kt = 0; kt < nk; ++kt) {
    const int k0 = kt * KB;
    __syncthreads();
#pragma unroll
    for (int p = 0; p < 4; ++p) {
      const int ra = (tid >> 3) + p * 32;
      const float4 f = *(const float4*)(A + (size_t)(m0 + ra) * K + k0 + (tid & 7) * 4);
      uint2 pk;
      pk.x = (unsigned)f2bf(f.x) | ((unsigned)f2bf(f.y) << 16);
      pk.y = (unsigned)f2bf(f.z) | ((unsigned)f2bf(f.w) << 16);
      *(uint2*)&smA[ra * LD + (tid & 7) * 4] = pk;
    }
#pragma unroll
    for (int p = 0; p < 2; ++p) {
      const int rb = (tid >> 2) + p * 64;
      const uint4 v = *(const uint4*)(Bt + (size_t)(n0 + rb) * K + k0 + (tid & 3) * 8);
      *(uint4*)&smB[rb * LD + (tid & 3) * 8] = v;
    }
    __syncthreads();
    compute_step(smA, smB, wm, wn, r, half, acc);
  }
  epilogue<MODE>(acc, bias, outv, M, Nc, m0, n0, wm, wn, r, half);
}

__global__ void k_cvt_bf16x8(const float* __restrict__ in, unsigned short* __restrict__ out,
                             size_t n8) {
  size_t i = (size_t)blockIdx.x * blockDim.x + threadIdx.x;
  if (i >= n8) return;
  const float4 f0 = *(const float4*)(in + i * 8);
  const float4 f1 = *(const float4*)(in + i * 8 + 4);
  Pack8 p;
  p.s[0] = f2bf(f0.x); p.s[1] = f2bf(f0.y); p.s[2] = f2bf(f0.z); p.s[3] = f2bf(f0.w);
  p.s[4] = f2bf(f1.x); p.s[5] = f2bf(f1.y); p.s[6] = f2bf(f1.z); p.s[7] = f2bf(f1.w);
  *(uint4*)(out + i * 8) = p.q;
}

// W [Kdim x Ncdim] f32 row-major -> Wt [Ncdim x Kdim] bf16
__global__ void k_transpose_bf16(const float* __restrict__ W, unsigned short* __restrict__ Wt,
                                 int Kdim, int Ncdim) {
  int i = blockIdx.x * blockDim.x + threadIdx.x;
  if (i < Kdim * Ncdim) {
    int k = i / Ncdim, n = i % Ncdim;
    Wt[(size_t)n * Kdim + k] = f2bf(W[i]);
  }
}

extern "C" void kernel_launch(void* const* d_in, const int* in_sizes, int n_in,
                              void* d_out, int out_size, void* d_ws, size_t ws_size,
                              hipStream_t stream)
{
  const float* x    = (const float*)d_in[0]; // [N, 512]
  const float* adj  = (const float*)d_in[1]; // [N, N]
  const float* W0   = (const float*)d_in[2]; // [512, 512]
  const float* b0   = (const float*)d_in[3]; // [512]
  const float* Wh   = (const float*)d_in[4]; // [2, 512, 512]
  const float* bh   = (const float*)d_in[5]; // [2, 512]
  const float* Wout = (const float*)d_in[6]; // [512, 128]
  const float* bout = (const float*)d_in[7]; // [128]
  float* out = (float*)d_out;                // [N, 128]

  unsigned short* ws    = (unsigned short*)d_ws;
  unsigned short* hbuf  = ws;                              // bf16 [N][512]
  unsigned short* suppT = hbuf  + (size_t)NROWS * HID;     // bf16 [512][N]
  unsigned short* W0t   = suppT + (size_t)HID * NROWS;     // bf16 [512][512]
  unsigned short* Wh0t  = W0t   + (size_t)HID * HID;
  unsigned short* Wh1t  = Wh0t  + (size_t)HID * HID;
  unsigned short* Woutt = Wh1t  + (size_t)HID * HID;       // bf16 [128][512]
  unsigned short* adjB  = Woutt + (size_t)NOUTC * HID;     // bf16 [N][N] (optional)

  const size_t need_full_bytes =
      ((size_t)(adjB - ws) + (size_t)NROWS * NROWS) * sizeof(unsigned short);
  const bool use_adjB = ws_size >= need_full_bytes;

  dim3 blk(256);
  k_cvt_bf16x8<<<(unsigned)(((size_t)NROWS * FEAT / 8 + 255) / 256), blk, 0, stream>>>(
      x, hbuf, (size_t)NROWS * FEAT / 8);
  k_transpose_bf16<<<(HID * HID + 255) / 256, blk, 0, stream>>>(W0, W0t, HID, HID);
  k_transpose_bf16<<<(HID * HID + 255) / 256, blk, 0, stream>>>(Wh, Wh0t, HID, HID);
  k_transpose_bf16<<<(HID * HID + 255) / 256, blk, 0, stream>>>(Wh + (size_t)HID * HID, Wh1t, HID, HID);
  k_transpose_bf16<<<(HID * NOUTC + 255) / 256, blk, 0, stream>>>(Wout, Woutt, HID, NOUTC);
  if (use_adjB)
    k_cvt_bf16x8<<<(unsigned)(((size_t)NROWS * NROWS / 8 + 255) / 256), blk, 0, stream>>>(
        adj, adjB, (size_t)NROWS * NROWS / 8);

  dim3 gSmall(NROWS / 128, HID / 128);   // 128 x 4
  dim3 gBig  (NROWS / 128, HID / 128);   // 128 x 4
  dim3 gOut  (NROWS / 128, NOUTC / 128); // 128 x 1

  const unsigned short* Wt_list[3] = { W0t, Wh0t, Wh1t };
  const float*          b_list[3]  = { b0, bh, bh + HID };

  for (int layer = 0; layer < 3; ++layer) {
    // suppT = (h @ W)^T
    gemm_wmma_async<0><<<gSmall, blk, 0, stream>>>(
        hbuf, Wt_list[layer], nullptr, suppT, NROWS, HID, HID);
    // h = relu(adj @ supp + b)
    if (use_adjB)
      gemm_wmma_async<1><<<gBig, blk, 0, stream>>>(
          adjB, suppT, b_list[layer], hbuf, NROWS, HID, NROWS);
    else
      gemm_wmma_f32A<1><<<gBig, blk, 0, stream>>>(
          adj, suppT, b_list[layer], hbuf, NROWS, HID, NROWS);
  }

  // out = h @ Wout + bout (f32)
  gemm_wmma_async<2><<<gOut, blk, 0, stream>>>(hbuf, Woutt, bout, out, NROWS, NOUTC, HID);
}
